// ResidualSimVQ_8495445311896
// MI455X (gfx1250) — matrix operand order repaired and checked by reading, not compiled, not run on hardware
//
#include <hip/hip_runtime.h>
#include <hip/hip_bf16.h>
#include <math.h>
#include <stdint.h>

// ---------------------------------------------------------------------------
// Residual SimVQ forward for MI455X (gfx1250, wave32, WMMA + TDM).
// Dominant cost: 4 x [16384 x 4096 x 512] distance GEMM -> bf16-split WMMA
// (hi*hi + hi*lo + lo*hi, f32 accumulate) on v_wmma_f32_16x16x32_bf16.
// A-tiles are DMA'd into LDS with tensor_load_to_lds (TENSORcnt).
// ---------------------------------------------------------------------------

typedef __attribute__((ext_vector_type(16))) __bf16 v16bf;
typedef __attribute__((ext_vector_type(8)))  float  v8f;

#define WMMA_BF16(A, B, C)                                                    \
  __builtin_amdgcn_wmma_f32_16x16x32_bf16(false, (A), false, (B), (short)0,   \
                                          (C), false, false)

#ifdef __has_builtin
#if __has_builtin(__builtin_amdgcn_tensor_load_to_lds) &&                     \
    __has_builtin(__builtin_amdgcn_s_wait_tensorcnt)
#define HAVE_TDM 1
#endif
#endif

static constexpr int   Bb   = 8;
static constexpr int   Nn   = 2048;
static constexpr int   Dd   = 512;
static constexpr int   Cc   = 4096;
static constexpr int   Qq   = 4;
static constexpr int   Mm   = Bb * Nn;          // 16384 rows
static constexpr long  QOUT = (long)Mm * Dd;    // 8388608 floats
static constexpr long  IDXB = QOUT;             // indices base in d_out
static constexpr long  LOSSB = QOUT + (long)Mm * Qq;
static constexpr float EPSf  = 1e-6f;
static constexpr float CIW   = 0.25f;           // commit input weight

// --- TDM: async DMA of a 2-D fp32 tile (rows x 512, stride 512) into LDS ---
#ifdef HAVE_TDM
__device__ __forceinline__ void tdm_load_tile_f32(const float* gsrc,
                                                  void* lds_dst, int rows) {
  typedef unsigned int u32x4 __attribute__((ext_vector_type(4)));
  typedef int          i32x4 __attribute__((ext_vector_type(4)));
  typedef int          i32x8 __attribute__((ext_vector_type(8)));
  const uint64_t ga = (uint64_t)(uintptr_t)gsrc;
  const uint32_t la = (uint32_t)(uintptr_t)lds_dst;  // low 32b = LDS offset
  u32x4 g0;
  g0[0] = 1u;                                   // count=1, user descriptor
  g0[1] = la;                                   // lds_addr (bytes)
  g0[2] = (uint32_t)ga;                         // global_addr[31:0]
  g0[3] = (uint32_t)(ga >> 32) | (2u << 30);    // global_addr[56:32] | type=2
  i32x8 g1;
  g1[0] = (int)(2u << 16);                      // data_size = 4 bytes
  g1[1] = (int)(512u << 16);                    // tensor_dim0 = 512
  g1[2] = (int)((uint32_t)rows << 16);          // tensor_dim1 = rows
  g1[3] = (int)(512u << 16);                    // tile_dim0 = 512
  g1[4] = rows;                                 // tile_dim1 = rows
  g1[5] = 512;                                  // tensor_dim0_stride = 512
  g1[6] = 0;
  g1[7] = 0;
  i32x4 z4;
  z4[0] = z4[1] = z4[2] = z4[3] = 0;            // groups 2/3 unused (2-D)
#if __clang_major__ >= 23
  i32x8 z8;
#pragma unroll
  for (int i = 0; i < 8; ++i) z8[i] = 0;
  __builtin_amdgcn_tensor_load_to_lds(g0, g1, z4, z4, z8, 0);
#else
  __builtin_amdgcn_tensor_load_to_lds(g0, g1, z4, z4, 0);
#endif
}
#endif

// --- A-operand fragment layout used here (contiguous k, B matches) --------------
// lane<16 : row M=lane,    K {tK+0..15}
// lane>=16: row M=lane-16, K {tK+16..31}
// Stored in LDS so each lane's 16 bf16 values are contiguous (32B aligned).
__device__ __forceinline__ void fill_afrags(__bf16* shi, __bf16* slo,
                                            const float* __restrict__ src,
                                            int ldsrc) {
  for (int idx = threadIdx.x; idx < 16 * Dd; idx += blockDim.x) {
    const int r = idx >> 9;       // row within 16-row tile
    const int k = idx & 511;      // K index
    const float v  = src[r * ldsrc + k];
    const __bf16 h = (__bf16)v;
    const __bf16 lo = (__bf16)(v - (float)h);
    const int t    = k >> 5;              // ktile
    const int kk   = k & 31;
    const int g    = kk >> 4;             // lane group
    const int j    = kk & 15;
    const int lane = g * 16 + r;
    const int slot = (t * 32 + lane) * 16 + j;
    shi[slot] = h;
    slo[slot] = lo;
  }
}

// Stage a 16x512 fp32 tile into LDS (TDM if available), then build fragments.
__device__ __forceinline__ void stage_and_split(__bf16* shi, __bf16* slo,
                                                float* stage,
                                                const float* __restrict__ gsrc) {
#ifdef HAVE_TDM
  if (threadIdx.x < 32) {                 // wave 0 issues the DMA and waits
    tdm_load_tile_f32(gsrc, stage, 16);
    __builtin_amdgcn_s_wait_tensorcnt(0);
  }
  __syncthreads();                        // publish LDS tile to all waves
  fill_afrags(shi, slo, stage, Dd);
#else
  (void)stage;
  fill_afrags(shi, slo, gsrc, Dd);
#endif
  __syncthreads();
}

// --- init: zero d_out (poisoned), residual = x, zero loss accumulators -----
__global__ void init_kernel(const float* __restrict__ x, float* __restrict__ out,
                            int out_n, float* __restrict__ res,
                            float* __restrict__ acc) {
  const long i = (long)blockIdx.x * blockDim.x + threadIdx.x;
  if (i < out_n) out[i] = 0.0f;
  if (i < QOUT) res[i] = x[i];
  if (i < 2 * Qq) acc[i] = 0.0f;
}

// --- stage 1: codebook = frozen @ W^T, fp32 + bf16 hi/lo splits ------------
// out[c,d] = sum_k F[c,k] * W[d,k] ; both operands K-contiguous.
__global__ __launch_bounds__(256) void transform_kernel(
    const float* __restrict__ fcb, const float* __restrict__ Wm,
    float* __restrict__ cb, __bf16* __restrict__ cbhi,
    __bf16* __restrict__ cblo) {
  __shared__ __align__(32) __bf16 sAhi[16 * Dd];
  __shared__ __align__(32) __bf16 sAlo[16 * Dd];
  __shared__ __align__(16) float  sStage[16 * Dd];
  const int c0 = blockIdx.x * 16;    // code-row tile
  const int d0 = blockIdx.y * 128;   // output-col stripe
  stage_and_split(sAhi, sAlo, sStage, fcb + (long)c0 * Dd);

  const int wv = threadIdx.x >> 5;
  const int l  = threadIdx.x & 31;
  const int lg = l >> 4;
  const int ln = l & 15;
  const int dcol = d0 + wv * 16 + ln;   // this lane's B column

  v8f acc = {};
  for (int t = 0; t < 16; ++t) {
    const v16bf ahi = *(const v16bf*)&sAhi[(t * 32 + l) * 16];
    const v16bf alo = *(const v16bf*)&sAlo[(t * 32 + l) * 16];
    const float* wp = Wm + (long)dcol * Dd + t * 32 + lg * 16;
    v16bf bhi, blo;
#pragma unroll
    for (int jj = 0; jj < 16; ++jj) {
      const float v  = wp[jj];
      const __bf16 h = (__bf16)v;
      bhi[jj] = h;
      blo[jj] = (__bf16)(v - (float)h);
    }
    acc = WMMA_BF16(ahi, bhi, acc);
    acc = WMMA_BF16(ahi, blo, acc);
    acc = WMMA_BF16(alo, bhi, acc);
  }
#pragma unroll
  for (int j = 0; j < 8; ++j) {
    const int  c = c0 + j + lg * 8;
    const long o = (long)c * Dd + dcol;
    const float v  = acc[j];
    const __bf16 h = (__bf16)v;
    cb[o]   = v;
    cbhi[o] = h;
    cblo[o] = (__bf16)(v - (float)h);
  }
}

// --- stage 2: code-row squared norms ---------------------------------------
__global__ __launch_bounds__(256) void cnorm_kernel(const float* __restrict__ cb,
                                                    float* __restrict__ cn) {
  const int c = blockIdx.x;
  const float* row = cb + (long)c * Dd;
  float s = 0.0f;
  for (int d = threadIdx.x; d < Dd; d += 256) {
    const float v = row[d];
    s += v * v;
  }
  for (int off = 16; off; off >>= 1) s += __shfl_xor(s, off, 32);
  __shared__ float red[8];
  if ((threadIdx.x & 31) == 0) red[threadIdx.x >> 5] = s;
  __syncthreads();
  if (threadIdx.x == 0) {
    float t = 0.0f;
    for (int i = 0; i < 8; ++i) t += red[i];
    cn[c] = t;
  }
}

// --- stage 3: distance GEMM + argmin (the compute hot spot) ----------------
// One WG = 16 residual rows x all 4096 codes. score = cnorm(c) - 2*<x,cb_c>.
__global__ __launch_bounds__(256) void argmin_kernel(
    const float* __restrict__ res, const __bf16* __restrict__ cbhi,
    const __bf16* __restrict__ cblo, const float* __restrict__ cn,
    int* __restrict__ idxout) {
  __shared__ __align__(32) __bf16 sAhi[16 * Dd];
  __shared__ __align__(32) __bf16 sAlo[16 * Dd];
  __shared__ __align__(16) float  sStage[16 * Dd];
  __shared__ float redv[8][16];
  __shared__ int   redi[8][16];
  const int m0 = blockIdx.x * 16;
  stage_and_split(sAhi, sAlo, sStage, res + (long)m0 * Dd);

  const int wv = threadIdx.x >> 5;
  const int l  = threadIdx.x & 31;
  const int lg = l >> 4;
  const int ln = l & 15;

  float minv[8];
  int   mini[8];
#pragma unroll
  for (int j = 0; j < 8; ++j) { minv[j] = 3.402823466e38f; mini[j] = 0; }

  for (int ch = 0; ch < 8; ++ch) {            // 8 chunks x 64 codes per wave
    const int cbase = wv * 512 + ch * 64;
    // warm L2/L0 for this chunk's codebook rows
    __builtin_prefetch(&cbhi[(long)cbase * Dd], 0, 0);
    __builtin_prefetch(&cblo[(long)cbase * Dd], 0, 0);
    v8f acc[4];
#pragma unroll
    for (int i = 0; i < 4; ++i) { v8f z = {}; acc[i] = z; }
    for (int t = 0; t < 16; ++t) {            // K = 512 in steps of 32
      const v16bf ahi = *(const v16bf*)&sAhi[(t * 32 + l) * 16];
      const v16bf alo = *(const v16bf*)&sAlo[(t * 32 + l) * 16];
#pragma unroll
      for (int i = 0; i < 4; ++i) {
        const int  c   = cbase + i * 16 + ln;
        const long off = (long)c * Dd + t * 32 + lg * 16;
        const v16bf bhi = *(const v16bf*)&cbhi[off];
        const v16bf blo = *(const v16bf*)&cblo[off];
        acc[i] = WMMA_BF16(ahi, bhi, acc[i]);
        acc[i] = WMMA_BF16(ahi, blo, acc[i]);
        acc[i] = WMMA_BF16(alo, bhi, acc[i]);
      }
    }
#pragma unroll
    for (int i = 0; i < 4; ++i) {
      const int   c   = cbase + i * 16 + ln;
      const float cnv = cn[c];
#pragma unroll
      for (int j = 0; j < 8; ++j) {           // lane holds M = j + lg*8
        const float s = cnv - 2.0f * acc[i][j];
        if (s < minv[j] || (s == minv[j] && c < mini[j])) {
          minv[j] = s;
          mini[j] = c;
        }
      }
    }
  }
  // reduce across the 16 lanes of each lane-group (same M, different codes)
#pragma unroll
  for (int j = 0; j < 8; ++j) {
#pragma unroll
    for (int off = 1; off < 16; off <<= 1) {
      const float ov = __shfl_xor(minv[j], off, 32);
      const int   oi = __shfl_xor(mini[j], off, 32);
      if (ov < minv[j] || (ov == minv[j] && oi < mini[j])) {
        minv[j] = ov;
        mini[j] = oi;
      }
    }
  }
  if (ln == 0) {
#pragma unroll
    for (int j = 0; j < 8; ++j) {
      redv[wv][lg * 8 + j] = minv[j];
      redi[wv][lg * 8 + j] = mini[j];
    }
  }
  __syncthreads();
  if (threadIdx.x < 16) {                      // combine the 8 waves' ranges
    float bv = redv[0][threadIdx.x];
    int   bi = redi[0][threadIdx.x];
    for (int w2 = 1; w2 < 8; ++w2) {
      const float v  = redv[w2][threadIdx.x];
      const int   i2 = redi[w2][threadIdx.x];
      if (v < bv || (v == bv && i2 < bi)) { bv = v; bi = i2; }
    }
    idxout[m0 + threadIdx.x] = bi;
  }
}

// --- stage 4: rotation-trick STE value, residual update, losses, indices ---
__global__ __launch_bounds__(256) void rotate_kernel(
    float* __restrict__ res, const float* __restrict__ cb,
    const int* __restrict__ idxin, float* __restrict__ out,
    float* __restrict__ acc, int qi) {
  const int m   = blockIdx.x;
  const int tid = threadIdx.x;
  const int idx = idxin[m];
  const float* e = res + (long)m * Dd;
  const float* tt = cb + (long)idx * Dd;
  const float e0 = e[tid], e1 = e[tid + 256];
  const float t0 = tt[tid], t1 = tt[tid + 256];

  float se2 = e0 * e0 + e1 * e1;
  float st2 = t0 * t0 + t1 * t1;
  float set = e0 * t0 + e1 * t1;
  for (int off = 16; off; off >>= 1) {
    se2 += __shfl_xor(se2, off, 32);
    st2 += __shfl_xor(st2, off, 32);
    set += __shfl_xor(set, off, 32);
  }
  __shared__ float r0[8], r1[8], r2[8];
  if ((tid & 31) == 0) { r0[tid >> 5] = se2; r1[tid >> 5] = st2; r2[tid >> 5] = set; }
  __syncthreads();
  __shared__ float sc[6];
  if (tid == 0) {
    float a2 = 0.0f, b2 = 0.0f, ab = 0.0f;
    for (int i = 0; i < 8; ++i) { a2 += r0[i]; b2 += r1[i]; ab += r2[i]; }
    const float nsrc = sqrtf(a2), ntgt = sqrtf(b2);
    const float a = fmaxf(nsrc, EPSf), b = fmaxf(ntgt, EPSf);
    const float nw = sqrtf(a2 / (a * a) + 2.0f * ab / (a * b) + b2 / (b * b));
    const float c = fmaxf(nw, EPSf);
    sc[0] = a; sc[1] = b; sc[2] = c;
    sc[3] = (a2 / a + ab / b) / c;  // <e, w>
    sc[4] = a2 / a;                 // <e, u>
    sc[5] = ntgt / a;               // norm ratio scale
  }
  __syncthreads();
  const float a = sc[0], b = sc[1], c = sc[2];
  const float dew = sc[3], deu = sc[4], scl = sc[5];

  float l1 = 0.0f, l2 = 0.0f;
#pragma unroll
  for (int p = 0; p < 2; ++p) {
    const int   d  = tid + p * 256;
    const float ev = p ? e1 : e0;
    const float tv = p ? t1 : t0;
    const float qd  = tv / b;
    const float wd  = (ev / a + qd) / c;
    const float rot = (ev - 2.0f * dew * wd + 2.0f * deu * qd) * scl;
    l1 += (ev - tv) * (ev - tv);
    const float dr = ev - rot;
    l2 += dr * dr;
    res[(long)m * Dd + d] = ev - rot;   // residual -= sg(quantized)
    out[(long)m * Dd + d] += rot;       // quantized_out += quantized
  }
  for (int off = 16; off; off >>= 1) {
    l1 += __shfl_xor(l1, off, 32);
    l2 += __shfl_xor(l2, off, 32);
  }
  __shared__ float s1[8], s2[8];
  if ((tid & 31) == 0) { s1[tid >> 5] = l1; s2[tid >> 5] = l2; }
  __syncthreads();
  if (tid == 0) {
    float a1 = 0.0f, a2s = 0.0f;
    for (int i = 0; i < 8; ++i) { a1 += s1[i]; a2s += s2[i]; }
    atomicAdd(&acc[qi * 2],     a1);
    atomicAdd(&acc[qi * 2 + 1], a2s);
    out[IDXB + (long)m * Qq + qi] = (float)idx;
  }
}

// --- finalize: loss means --------------------------------------------------
__global__ void finalize_kernel(const float* __restrict__ acc,
                                float* __restrict__ out) {
  const int q = threadIdx.x;
  if (q < Qq) {
    const float inv = 1.0f / (float)((long)Mm * Dd);
    out[LOSSB + q] = acc[q * 2] * inv + CIW * (acc[q * 2 + 1] * inv);
  }
}

// ---------------------------------------------------------------------------
extern "C" void kernel_launch(void* const* d_in, const int* in_sizes, int n_in,
                              void* d_out, int out_size, void* d_ws,
                              size_t ws_size, hipStream_t stream) {
  const float* x   = (const float*)d_in[0];   // [8,2048,512]
  const float* fcb = (const float*)d_in[1];   // [4,4096,512]
  const float* Wts = (const float*)d_in[2];   // [4,512,512]
  float* out = (float*)d_out;
  char*  ws  = (char*)d_ws;

  // workspace layout (all 256B aligned)
  float*  res  = (float*)(ws + 0);           // 32 MB residual
  float*  cb   = (float*)(ws + 33554432);    //  8 MB fp32 codebook
  __bf16* cbhi = (__bf16*)(ws + 41943040);   //  4 MB bf16 hi
  __bf16* cblo = (__bf16*)(ws + 45088768);   //  4 MB bf16 lo
  float*  cn   = (float*)(ws + 49283072);    // 16 KB code norms
  int*    idxb = (int*)(ws + 49299456);      // 64 KB indices
  float*  acc  = (float*)(ws + 49365504);    // loss accumulators (8 floats)

  const long tot = (out_size > (int)QOUT) ? (long)out_size : QOUT;
  init_kernel<<<(int)((tot + 255) / 256), 256, 0, stream>>>(x, out, out_size,
                                                            res, acc);
  for (int qi = 0; qi < Qq; ++qi) {
    transform_kernel<<<dim3(Cc / 16, Dd / 128), 256, 0, stream>>>(
        fcb + (long)qi * Cc * Dd, Wts + (long)qi * Dd * Dd, cb, cbhi, cblo);
    cnorm_kernel<<<Cc, 256, 0, stream>>>(cb, cn);
    argmin_kernel<<<Mm / 16, 256, 0, stream>>>(res, cbhi, cblo, cn, idxb);
    rotate_kernel<<<Mm, 256, 0, stream>>>(res, cb, idxb, out, acc, qi);
  }
  finalize_kernel<<<1, 32, 0, stream>>>(acc, out);
}
